// LSTMCell_61890478735973
// MI455X (gfx1250) — compile-verified, matrix operand order checked by reading, not run on hardware
//
#include <hip/hip_runtime.h>

#define T_STEPS 256
#define BATCH   64
#define IN_DIM  1024
#define HID     1024
#define H4      4096
#define MROWS   (T_STEPS * BATCH)   // 16384
#define NBLK    (HID / 16)          // 64 persistent blocks

typedef __attribute__((ext_vector_type(16))) __bf16         v16bf;
typedef __attribute__((ext_vector_type(8)))  float          v8f;
typedef __attribute__((ext_vector_type(8)))  unsigned short v8u16;

// fp32 -> bf16 round-to-nearest-even (bits kept as ushort: host-safe pointer type)
__device__ __forceinline__ unsigned short f2bf_bits(float f) {
    unsigned u = __builtin_bit_cast(unsigned, f);
    u += 0x7FFFu + ((u >> 16) & 1u);
    return (unsigned short)(u >> 16);
}

// 16x32 bf16 A/B fragment per CDNA5 ISA 7.12.2:
// lane l: rc=l&15 (row for A / col-as-row for B), half=l>>4.
// elements 0..7  <- K = k0 + half*8 .. +7      (16 contiguous bytes)
// elements 8..15 <- K = k0 + 16 + half*8 .. +7 (16 contiguous bytes)
__device__ __forceinline__ v16bf load_frag16x32(const unsigned short* __restrict__ base,
                                                int ld, int rowBase, int k0, int lane) {
    const int rc = lane & 15, half = (lane >> 4) & 1;
    const unsigned short* p = base + (size_t)(rowBase + rc) * ld + k0 + half * 8;
    v8u16 lo = *(const v8u16*)p;
    v8u16 hi = *(const v8u16*)(p + 16);
    v16bf r;
#pragma unroll
    for (int i = 0; i < 8; ++i) {
        r[i]     = __builtin_bit_cast(__bf16, (unsigned short)lo[i]);
        r[i + 8] = __builtin_bit_cast(__bf16, (unsigned short)hi[i]);
    }
    return r;
}

__device__ __forceinline__ v8f wmma_bf16(v16bf a, v16bf b, v8f c) {
    return __builtin_amdgcn_wmma_f32_16x16x32_bf16(false, a, false, b, (short)0, c,
                                                   false, false);
}

__device__ __forceinline__ float sigmoidf_fast(float x) {
    return 1.0f / (1.0f + __expf(-x));
}

// ---------------------------------------------------------------------------
// preact[M,4H] = Xb[M,I] @ Wih^T[I,4H] + (b_ih + b_hh)   (fully parallel)
// block: 256 thr = 8 waves; block tile 64 rows x 128 cols; wave = one 16-col strip
// ---------------------------------------------------------------------------
__global__ __launch_bounds__(256) void lstm_pre_gemm(
    const unsigned short* __restrict__ Xb,    // [MROWS, IN_DIM] bf16
    const unsigned short* __restrict__ Wib,   // [H4, IN_DIM]    bf16 (row = out col)
    const float* __restrict__ b_ih,
    const float* __restrict__ b_hh,
    const int*  __restrict__ biasFlag,
    float* __restrict__ preact)               // [MROWS, H4] fp32
{
    const int lane    = threadIdx.x & 31;
    const int wave    = threadIdx.x >> 5;               // 0..7
    const int colBase = blockIdx.x * 128 + wave * 16;   // N in [0, 4096)
    const int rowBase = blockIdx.y * 64;                // M in [0, 16384)

    v8f acc[4] = {};
    for (int k = 0; k < IN_DIM; k += 32) {
        v16bf bfrag = load_frag16x32(Wib, IN_DIM, colBase, k, lane);
#pragma unroll
        for (int mt = 0; mt < 4; ++mt) {
            v16bf afrag = load_frag16x32(Xb, IN_DIM, rowBase + mt * 16, k, lane);
            acc[mt] = wmma_bf16(afrag, bfrag, acc[mt]);
        }
    }

    const int rc = lane & 15, half = (lane >> 4) & 1;
    const int col = colBase + rc;
    const float bias = (*biasFlag) ? (b_ih[col] + b_hh[col]) : 0.0f;
#pragma unroll
    for (int mt = 0; mt < 4; ++mt) {
#pragma unroll
        for (int e = 0; e < 8; ++e) {
            const int m = rowBase + mt * 16 + half * 8 + e;
            preact[(size_t)m * H4 + col] = acc[mt][e] + bias;
        }
    }
}

// ---------------------------------------------------------------------------
// Persistent recurrent kernel: 64 blocks live for all 256 timesteps.
//   - Wave g owns gate g's 16-row strip of W_hh, held ENTIRELY in registers
//     (32 x v16bf = 256 VGPRs) for the whole recurrence.
//   - Per step: gates = preact_t + h @ Whh^T via 128 register-resident WMMA,
//     fused sigmoid/tanh cell update, h ping-pongs hA<->hB.
//   - Steps separated by an agent-scope atomic grid barrier (monotone counter).
//   - Next step's preact tile prefetched (global_prefetch) under the WMMA loop.
// ---------------------------------------------------------------------------
__global__ __launch_bounds__(128, 1) void lstm_recurrent(
    const unsigned short* __restrict__ Whb,     // [H4, HID] bf16
    const float* __restrict__ preact,           // [MROWS, H4] fp32
    float* __restrict__ cstate,                 // [B, HID] fp32 (in/out)
    unsigned short* __restrict__ hA,            // [B, HID] bf16 (holds h_0)
    unsigned short* __restrict__ hB,            // [B, HID] bf16
    float* __restrict__ out,                    // [T, B, HID] fp32
    unsigned* __restrict__ barCtr)              // grid barrier counter (zeroed)
{
    __shared__ float lds[4][BATCH][16];         // 16 KB: i,f,g,o tiles

    const int lane = threadIdx.x & 31;
    const int g    = threadIdx.x >> 5;          // gate 0..3
    const int colH = blockIdx.x * 16;           // column tile in H
    const int wrowBase = g * HID + colH;        // Whh rows for this gate strip

    // ---- preload the whole W_hh strip into registers (32 K-fragments) ----
    v16bf wfrag[32];
#pragma unroll
    for (int kk = 0; kk < 32; ++kk)
        wfrag[kk] = load_frag16x32(Whb, HID, wrowBase, kk * 32, lane);

    const int rc = lane & 15, half = (lane >> 4) & 1;
    const int col4 = g * HID + colH + rc;       // column in [0, 4H)

    for (int t = 0; t < T_STEPS; ++t) {
        const unsigned short* hin  = (t & 1) ? hB : hA;
        unsigned short*       hout = (t & 1) ? hA : hB;
        const float* preact_t = preact + (size_t)t * BATCH * H4;

        // prefetch next step's preact tile (overlaps with the WMMA loop below)
        if (t + 1 < T_STEPS) {
            const float* pn = preact + (size_t)(t + 1) * BATCH * H4;
            const int m  = threadIdx.x & 63;        // 0..63 rows
            const int g0 = threadIdx.x >> 6;        // 0..1
            __builtin_prefetch(pn + (size_t)m * H4 + (g0 * HID + colH), 0, 1);
            __builtin_prefetch(pn + (size_t)m * H4 + ((g0 + 2) * HID + colH), 0, 1);
        }

        // ---- recurrent GEMM: register-resident weights ----
        v8f acc[4] = {};
#pragma unroll
        for (int kk = 0; kk < 32; ++kk) {
#pragma unroll
            for (int mt = 0; mt < 4; ++mt) {
                v16bf afrag = load_frag16x32(hin, HID, mt * 16, kk * 32, lane);
                acc[mt] = wmma_bf16(afrag, wfrag[kk], acc[mt]);
            }
        }

        // ---- gate tiles -> LDS (add preact) ----
#pragma unroll
        for (int mt = 0; mt < 4; ++mt) {
#pragma unroll
            for (int e = 0; e < 8; ++e) {
                const int m = mt * 16 + half * 8 + e;
                lds[g][m][rc] = acc[mt][e] + preact_t[(size_t)m * H4 + col4];
            }
        }
        __syncthreads();

        // ---- fused elementwise LSTM update: 64x16 elems over 128 threads ----
        float* out_t = out + (size_t)t * BATCH * HID;
#pragma unroll
        for (int j = 0; j < 8; ++j) {
            const int idx = j * 128 + threadIdx.x;  // 0..1023
            const int m = idx >> 4, n = idx & 15;
            const int col = colH + n;
            const float ig = lds[0][m][n];
            const float fg = lds[1][m][n];
            const float gg = lds[2][m][n];
            const float og = lds[3][m][n];
            const float cp = cstate[m * HID + col];
            const float cn = sigmoidf_fast(fg) * cp + sigmoidf_fast(ig) * tanhf(gg);
            const float hn = sigmoidf_fast(og) * tanhf(cn);
            cstate[m * HID + col]  = cn;
            out_t[m * HID + col]   = hn;
            hout[m * HID + col]    = f2bf_bits(hn);
        }
        __syncthreads();

        // ---- agent-scope grid barrier (monotone counter, no reset race) ----
        if (threadIdx.x == 0) {
            __hip_atomic_fetch_add(barCtr, 1u, __ATOMIC_ACQ_REL,
                                   __HIP_MEMORY_SCOPE_AGENT);
            const unsigned target = (unsigned)(t + 1) * (unsigned)gridDim.x;
            while (__hip_atomic_load(barCtr, __ATOMIC_ACQUIRE,
                                     __HIP_MEMORY_SCOPE_AGENT) < target)
                __builtin_amdgcn_s_sleep(1);
        }
        __syncthreads();
    }
}

// ---------------------------------------------------------------------------
// Helpers: conversion / init / finalize
// ---------------------------------------------------------------------------
__global__ void f32_to_bf16(const float* __restrict__ src,
                            unsigned short* __restrict__ dst, size_t n) {
    size_t i = (size_t)blockIdx.x * blockDim.x + threadIdx.x;
    const size_t stride = (size_t)gridDim.x * blockDim.x;
    for (; i < n; i += stride) dst[i] = f2bf_bits(src[i]);
}

__global__ void lstm_init(const float* __restrict__ hidden,
                          const float* __restrict__ cand,
                          unsigned short* __restrict__ hbf,
                          float* __restrict__ cstate,
                          unsigned* __restrict__ barCtr, int n) {
    const int i = blockIdx.x * blockDim.x + threadIdx.x;
    if (i == 0) *barCtr = 0u;                 // reset grid barrier every launch
    if (i < n) {
        hbf[i]    = f2bf_bits(hidden[i]);
        cstate[i] = cand[i];
    }
}

__global__ void lstm_finalize(const float* __restrict__ h_last,
                              const float* __restrict__ cstate,
                              float* __restrict__ dst_h,
                              float* __restrict__ dst_c, int n) {
    const int i = blockIdx.x * blockDim.x + threadIdx.x;
    if (i < n) {
        dst_h[i] = h_last[i];
        dst_c[i] = cstate[i];
    }
}

// ---------------------------------------------------------------------------
extern "C" void kernel_launch(void* const* d_in, const int* in_sizes, int n_in,
                              void* d_out, int out_size, void* d_ws, size_t ws_size,
                              hipStream_t stream) {
    const float* x      = (const float*)d_in[0];   // [T,B,I]
    const float* hidden = (const float*)d_in[1];   // [B,H]
    const float* cand   = (const float*)d_in[2];   // [B,H]
    const float* w_ih   = (const float*)d_in[3];   // [4H,I]
    const float* w_hh   = (const float*)d_in[4];   // [4H,H]
    const float* b_ih   = (const float*)d_in[5];   // [4H]
    const float* b_hh   = (const float*)d_in[6];   // [4H]
    const int*   biasF  = (const int*)d_in[7];     // scalar flag
    float* out = (float*)d_out;                    // [T,B,H] + [B,H] + [B,H]

    char* p = (char*)d_ws;
    unsigned short* xb  = (unsigned short*)p; p += (size_t)MROWS * IN_DIM * 2; // 32 MB
    unsigned short* wib = (unsigned short*)p; p += (size_t)H4 * IN_DIM * 2;    //  8 MB
    unsigned short* whb = (unsigned short*)p; p += (size_t)H4 * HID * 2;       //  8 MB
    float* preact = (float*)p;                p += (size_t)MROWS * H4 * 4;     // 256 MB
    float* cstate = (float*)p;                p += (size_t)BATCH * HID * 4;
    unsigned short* hA = (unsigned short*)p;  p += (size_t)BATCH * HID * 2;
    unsigned short* hB = (unsigned short*)p;  p += (size_t)BATCH * HID * 2;
    unsigned* barCtr   = (unsigned*)p;        p += 256;

    // 1) one-time conversions to bf16 + state/barrier init
    f32_to_bf16<<<4096, 256, 0, stream>>>(x,    xb,  (size_t)MROWS * IN_DIM);
    f32_to_bf16<<<2048, 256, 0, stream>>>(w_ih, wib, (size_t)H4 * IN_DIM);
    f32_to_bf16<<<2048, 256, 0, stream>>>(w_hh, whb, (size_t)H4 * HID);
    lstm_init<<<(BATCH * HID + 255) / 256, 256, 0, stream>>>(hidden, cand, hA, cstate,
                                                             barCtr, BATCH * HID);

    // 2) hoisted input GEMM: preact = X @ Wih^T + b  (137 GFLOP, parallel)
    lstm_pre_gemm<<<dim3(H4 / 128, MROWS / 64), 256, 0, stream>>>(xb, wib, b_ih, b_hh,
                                                                  biasF, preact);

    // 3) persistent recurrence: one launch, 256 steps, grid-barrier per step
    lstm_recurrent<<<NBLK, 128, 0, stream>>>(whb, preact, cstate, hA, hB, out, barCtr);

    // 4) h_final = output[T-1]; c_final = cstate
    lstm_finalize<<<(BATCH * HID + 255) / 256, 256, 0, stream>>>(
        out + (size_t)(T_STEPS - 1) * BATCH * HID, cstate,
        out + (size_t)T_STEPS * BATCH * HID,
        out + (size_t)T_STEPS * BATCH * HID + (size_t)BATCH * HID,
        BATCH * HID);
}
